// GNN_v8_5927054868951
// MI455X (gfx1250) — matrix operand order repaired and checked by reading, not compiled, not run on hardware
//
#include <hip/hip_runtime.h>
#include <hip/hip_bf16.h>

// ---------------------------------------------------------------------------
// Problem constants (from reference)
// ---------------------------------------------------------------------------
#define NNODES   16384
#define NGRAPHS  256
#define NPG      64          // nodes per graph
#define NEDGES   262144
#define EPG      1024        // edges per graph (block-diagonal)
#define EDG      (EPG + NPG) // edges incl. self loops per graph = 1088
#define TOTEDGE  (NEDGES + NNODES)

typedef __attribute__((ext_vector_type(8)))  float  v8f;
typedef __attribute__((ext_vector_type(16))) __bf16 v16bf;

// ---------------------------------------------------------------------------
// WMMA GEMM:  C[M,N] = A[M,K] * B[K,N] (+bias[N])
// REQUIRES: M % 32 == 0, N % 16 == 0 (holds for every call site).
// fp32 in/out, bf16 WMMA. One wave computes a 32x16 output block (two 16x16
// tiles stacked in M) so the B fragment is reused by two v_wmma ops.
// A-frag (16-bit 16x32): lane m=lane&15, half=lane>>4; elem e maps to the two
//   contiguous 8-float runs [kb+half*8, +8) and [kb+16+half*8, +8)  -> float4s
// B-frag (32x16): col n=lane&15, k = kb + half*8 + (e&7) + 16*(e>>3) (coalesced across lanes)
// C/D: vgpr v -> row v + half*8, col lane&15
// f32 -> bf16 uses native fptrunc (RNE on gfx1250, bf16 is a legal VALU type).
// ---------------------------------------------------------------------------
__global__ __launch_bounds__(256)
void wmma_gemm_bias_kernel(const float* __restrict__ A, const float* __restrict__ B,
                           const float* __restrict__ bias, float* __restrict__ C,
                           int M, int N, int K, int tn) {
    int wave = threadIdx.x >> 5, lane = threadIdx.x & 31;
    int tile = blockIdx.x * 8 + wave;
    int tiles_total = (M >> 5) * tn;
    if (tile >= tiles_total) return;                 // wave-uniform
    int tm = tile / tn, tc = tile % tn;
    int half = lane >> 4, mr = lane & 15;
    int arow0 = tm * 32 + mr;                        // first M-tile row
    int bcol  = tc * 16 + mr;

    const float* Arow0 = A + (size_t)arow0 * K;
    const float* Arow1 = Arow0 + (size_t)16 * K;     // second M-tile row
    const float* Bcol  = B + bcol;

    v8f acc0 = {}, acc1 = {};
    union UA { v16bf v; __bf16 h[16]; };
    union AF { float4 q[4]; float f[16]; };

    const int kfull = K & ~31;
    for (int kb = 0; kb < kfull; kb += 32) {
        if (kb + 32 < kfull) {
            __builtin_prefetch(Arow0 + kb + 32, 0, 3);
            __builtin_prefetch(Arow1 + kb + 32, 0, 3);
        }
        AF af0, af1;
        // two contiguous 8-float segments per row, 16B aligned
        const float4* p00 = (const float4*)(Arow0 + kb + half * 8);
        const float4* p01 = (const float4*)(Arow0 + kb + 16 + half * 8);
        const float4* p10 = (const float4*)(Arow1 + kb + half * 8);
        const float4* p11 = (const float4*)(Arow1 + kb + 16 + half * 8);
        af0.q[0] = p00[0]; af0.q[1] = p00[1]; af0.q[2] = p01[0]; af0.q[3] = p01[1];
        af1.q[0] = p10[0]; af1.q[1] = p10[1]; af1.q[2] = p11[0]; af1.q[3] = p11[1];

        UA ua0, ua1, ub;
#pragma unroll
        for (int e = 0; e < 16; ++e) {
            ua0.h[e] = (__bf16)af0.f[e];
            ua1.h[e] = (__bf16)af1.f[e];
            ub.h[e]  = (__bf16)Bcol[(size_t)(kb + half * 8 + (e & 7) + ((e >> 3) << 4)) * N];
        }
        acc0 = __builtin_amdgcn_wmma_f32_16x16x32_bf16(false, ua0.v, false, ub.v,
                                                       (short)0, acc0, false, false);
        acc1 = __builtin_amdgcn_wmma_f32_16x16x32_bf16(false, ua1.v, false, ub.v,
                                                       (short)0, acc1, false, false);
    }

    if (kfull < K) {                                 // ragged K tail (K=3 case)
        UA ua0, ua1, ub;
#pragma unroll
        for (int e = 0; e < 16; ++e) {
            int v  = e >> 1;
            int ka = ((v >> 2) << 4) + (half << 3) + ((v & 3) << 1) + (e & 1);
            int k  = kfull + ka;
            ua0.h[e] = (k < K) ? (__bf16)Arow0[k] : (__bf16)0.f;
            ua1.h[e] = (k < K) ? (__bf16)Arow1[k] : (__bf16)0.f;
            int kbk = kfull + (half << 3) + (e & 7) + ((e >> 3) << 4);
            ub.h[e]  = (kbk < K) ? (__bf16)Bcol[(size_t)kbk * N] : (__bf16)0.f;
        }
        acc0 = __builtin_amdgcn_wmma_f32_16x16x32_bf16(false, ua0.v, false, ub.v,
                                                       (short)0, acc0, false, false);
        acc1 = __builtin_amdgcn_wmma_f32_16x16x32_bf16(false, ua1.v, false, ub.v,
                                                       (short)0, acc1, false, false);
    }

    float bv = bias ? bias[bcol] : 0.f;
#pragma unroll
    for (int v = 0; v < 8; ++v) {
        int row0 = tm * 32 + v + (half << 3);
        C[(size_t)row0 * N + bcol]        = acc0[v] + bv;
        C[(size_t)(row0 + 16) * N + bcol] = acc1[v] + bv;
    }
}

// ---------------------------------------------------------------------------
// GATv2 edge logits: e = sum_c att[c] * leakyrelu(xl[s][c] + xr[t][c], 0.2)
// One wave per edge (incl. self loops id>=E), lanes over channels.
// ---------------------------------------------------------------------------
__global__ __launch_bounds__(256)
void edge_logits_kernel(const float* __restrict__ XL, const float* __restrict__ XR,
                        const float* __restrict__ att, const int* __restrict__ src,
                        const int* __restrict__ tgt, float* __restrict__ logits, int C) {
    int wave = threadIdx.x >> 5, lane = threadIdx.x & 31;
    int eid = blockIdx.x * 8 + wave;
    if (eid >= TOTEDGE) return;                      // wave-uniform
    int s, t;
    if (eid < NEDGES) { s = src[eid]; t = tgt[eid]; }
    else              { s = t = eid - NEDGES; }
    float p = 0.f;
    for (int c = lane; c < C; c += 32) {
        float m = XL[(size_t)s * C + c] + XR[(size_t)t * C + c];
        m = m > 0.f ? m : 0.2f * m;
        p += m * att[c];
    }
    for (int off = 16; off > 0; off >>= 1) p += __shfl_down(p, off);
    if (lane == 0) logits[eid] = p;
}

// ---------------------------------------------------------------------------
// Segment softmax over incoming edges per target node (per-graph, all LDS).
// Overwrites logits with normalized alpha.
// ---------------------------------------------------------------------------
__global__ __launch_bounds__(256)
void softmax_stats_kernel(float* __restrict__ logits, const int* __restrict__ tgt) {
    int g = blockIdx.x, tid = threadIdx.x;
    __shared__ float ev[EDG];
    __shared__ unsigned char tl[EDG];
    __shared__ float mx[NPG], sm[NPG];
    for (int idx = tid; idx < EDG; idx += 256) {
        int eid, t;
        if (idx < EPG) { eid = g * EPG + idx; t = tgt[eid] - g * NPG; }
        else           { int n = idx - EPG; eid = NEDGES + g * NPG + n; t = n; }
        ev[idx] = logits[eid];
        tl[idx] = (unsigned char)t;
    }
    __syncthreads();
    if (tid < NPG) {
        float m = -3.0e38f;
        for (int i = 0; i < EDG; ++i) if (tl[i] == tid) m = fmaxf(m, ev[i]);
        mx[tid] = m;
        float s = 0.f;
        for (int i = 0; i < EDG; ++i) if (tl[i] == tid) s += __expf(ev[i] - m);
        sm[tid] = s;
    }
    __syncthreads();
    for (int idx = tid; idx < EDG; idx += 256) {
        int t = tl[idx];
        float a = __expf(ev[idx] - mx[t]) / sm[t];
        int eid = (idx < EPG) ? g * EPG + idx : NEDGES + g * NPG + (idx - EPG);
        logits[eid] = a;
    }
}

// ---------------------------------------------------------------------------
// GAT aggregation: out[t][c] = sum_e alpha_e * xl[s][c]  + bias[c]
// one block per (graph, 64-channel chunk); all accumulation via ds_add_f32.
// ---------------------------------------------------------------------------
__global__ __launch_bounds__(256)
void gat_aggregate_kernel(const float* __restrict__ XL, const float* __restrict__ alpha,
                          const int* __restrict__ src, const int* __restrict__ tgt,
                          const float* __restrict__ bias, float* __restrict__ out, int C) {
    int g = blockIdx.x, tid = threadIdx.x;
    int c0 = blockIdx.y * 64;
    int cw = min(64, C - c0);
    __shared__ float xls[NPG * 64], outs[NPG * 64];
    __shared__ float av[EDG];
    __shared__ unsigned char sl[EDG], tl[EDG];
    for (int idx = tid; idx < EDG; idx += 256) {
        int eid, s, t;
        if (idx < EPG) { eid = g * EPG + idx; s = src[eid] - g * NPG; t = tgt[eid] - g * NPG; }
        else           { int n = idx - EPG; eid = NEDGES + g * NPG + n; s = t = n; }
        av[idx] = alpha[eid]; sl[idx] = (unsigned char)s; tl[idx] = (unsigned char)t;
    }
    for (int i = tid; i < NPG * 64; i += 256) {
        outs[i] = 0.f;
        int n = i >> 6, c = i & 63;
        xls[i] = (c < cw) ? XL[((size_t)(g * NPG + n)) * C + c0 + c] : 0.f;
    }
    __syncthreads();
    for (int idx = tid; idx < EDG; idx += 256) {
        float a = av[idx]; int s = sl[idx], t = tl[idx];
        const float* xp = &xls[s * 64];
        float* op = &outs[t * 64];
        for (int c = 0; c < cw; ++c) atomicAdd(&op[c], a * xp[c]);
    }
    __syncthreads();
    for (int i = tid; i < NPG * 64; i += 256) {
        int n = i >> 6, c = i & 63;
        if (c < cw)
            out[((size_t)(g * NPG + n)) * C + c0 + c] = outs[i] + bias[c0 + c];
    }
}

// ---------------------------------------------------------------------------
// ARMA pieces: degree -> dinv (per-graph), then aggregation + relu
// ---------------------------------------------------------------------------
__global__ __launch_bounds__(256)
void deg_kernel(const int* __restrict__ tgt, float* __restrict__ dinv) {
    int g = blockIdx.x, tid = threadIdx.x;
    __shared__ int cnt[NPG];
    if (tid < NPG) cnt[tid] = 0;
    __syncthreads();
    for (int idx = tid; idx < EPG; idx += 256)
        atomicAdd(&cnt[tgt[g * EPG + idx] - g * NPG], 1);
    __syncthreads();
    if (tid < NPG) {
        int d = cnt[tid];
        dinv[g * NPG + tid] = (d > 0) ? rsqrtf((float)d) : 0.f;
    }
}

__global__ __launch_bounds__(256)
void arma_aggregate_kernel(const float* __restrict__ XI, const float* __restrict__ XR2,
                           const float* __restrict__ dinv, const int* __restrict__ src,
                           const int* __restrict__ tgt, float* __restrict__ out, int C) {
    int g = blockIdx.x, tid = threadIdx.x;
    int c0 = blockIdx.y * 64;
    int cw = min(64, C - c0);
    __shared__ float xis[NPG * 64], outs[NPG * 64];
    __shared__ float wv[EPG];
    __shared__ unsigned char sl[EPG], tl[EPG];
    __shared__ float dl[NPG];
    if (tid < NPG) dl[tid] = dinv[g * NPG + tid];
    __syncthreads();
    for (int idx = tid; idx < EPG; idx += 256) {
        int eid = g * EPG + idx;
        int s = src[eid] - g * NPG, t = tgt[eid] - g * NPG;
        sl[idx] = (unsigned char)s; tl[idx] = (unsigned char)t;
        wv[idx] = dl[s] * dl[t];
    }
    for (int i = tid; i < NPG * 64; i += 256) {
        outs[i] = 0.f;
        int n = i >> 6, c = i & 63;
        xis[i] = (c < cw) ? XI[((size_t)(g * NPG + n)) * C + c0 + c] : 0.f;
    }
    __syncthreads();
    for (int idx = tid; idx < EPG; idx += 256) {
        float w = wv[idx]; int s = sl[idx], t = tl[idx];
        const float* xp = &xis[s * 64];
        float* op = &outs[t * 64];
        for (int c = 0; c < cw; ++c) atomicAdd(&op[c], w * xp[c]);
    }
    __syncthreads();
    for (int i = tid; i < NPG * 64; i += 256) {
        int n = i >> 6, c = i & 63;
        if (c < cw) {
            size_t o = ((size_t)(g * NPG + n)) * C + c0 + c;
            out[o] = fmaxf(outs[i] + XR2[o], 0.f);   // relu(prop + xWr + b)
        }
    }
}

// ---------------------------------------------------------------------------
// Fused elu + GraphNorm.  thread = (graph, channel).  In-place safe.
// out[n] = w * (elu(x[n]) - mean*ms) / sqrt(var+eps) + b
// ---------------------------------------------------------------------------
__device__ __forceinline__ float elu1(float v) { return v > 0.f ? v : (__expf(v) - 1.f); }

__global__ __launch_bounds__(256)
void graph_norm_kernel(const float* __restrict__ in, float* __restrict__ out,
                       const float* __restrict__ w, const float* __restrict__ b,
                       const float* __restrict__ ms, int C, int ldo, int coff) {
    int g = blockIdx.x;
    int c = blockIdx.y * 256 + threadIdx.x;
    if (c >= C) return;
    const float* ip = in + (size_t)g * NPG * C + c;
    float s = 0.f;
    for (int n = 0; n < NPG; ++n) s += elu1(ip[(size_t)n * C]);
    float mean = s * (1.f / NPG);
    float msv = ms[c], vv = 0.f;
    for (int n = 0; n < NPG; ++n) {
        float d = elu1(ip[(size_t)n * C]) - mean * msv;
        vv += d * d;
    }
    float scale = w[c] * rsqrtf(vv * (1.f / NPG) + 1e-5f);
    float bb = b[c];
    float* op = out + (size_t)g * NPG * ldo + coff + c;
    for (int n = 0; n < NPG; ++n) {
        float d = elu1(ip[(size_t)n * C]) - mean * msv;
        op[(size_t)n * ldo] = d * scale + bb;
    }
}

// ---------------------------------------------------------------------------
// Pooling: max / mean / sum over 64 nodes per graph -> pooled[B, 3072]
// ---------------------------------------------------------------------------
__global__ __launch_bounds__(256)
void pool_kernel(const float* __restrict__ H, float* __restrict__ pooled) {
    int g = blockIdx.x;
    int c = blockIdx.y * 256 + threadIdx.x;
    if (c >= 1024) return;
    const float* hp = H + (size_t)g * NPG * 1024 + c;
    float mx = -3.0e38f, s = 0.f;
    for (int n = 0; n < NPG; ++n) {
        float v = hp[(size_t)n * 1024];
        mx = fmaxf(mx, v); s += v;
    }
    pooled[(size_t)g * 3072 + c]        = mx;
    pooled[(size_t)g * 3072 + 1024 + c] = s * (1.f / NPG);
    pooled[(size_t)g * 3072 + 2048 + c] = s;
}

// ---------------------------------------------------------------------------
// SortAggregation(k=4): rank-select top-4 by last channel (stable desc).
// ---------------------------------------------------------------------------
__global__ __launch_bounds__(256)
void sort_topk_kernel(const float* __restrict__ H, float* __restrict__ sorted) {
    int g = blockIdx.x, tid = threadIdx.x;
    __shared__ float key[NPG];
    __shared__ int sel[4];
    if (tid < NPG) key[tid] = H[((size_t)(g * NPG + tid)) * 1024 + 1023];
    __syncthreads();
    if (tid < NPG) {
        float vi = key[tid];
        int rank = 0;
        for (int j = 0; j < NPG; ++j) {
            float vj = key[j];
            rank += (vj > vi) || (vj == vi && j < tid);   // stable argsort(-x)
        }
        if (rank < 4) sel[rank] = tid;
    }
    __syncthreads();
    for (int i = tid; i < 4096; i += 256) {
        int r = i >> 10, c = i & 1023;
        sorted[(size_t)g * 4096 + i] = H[((size_t)(g * NPG + sel[r])) * 1024 + c];
    }
}

// ---------------------------------------------------------------------------
// af = additional_feat[B,9] @ lin4_W[9,128] + lin4_b
// ---------------------------------------------------------------------------
__global__ __launch_bounds__(128)
void lin4_kernel(const float* __restrict__ af_in, const float* __restrict__ W,
                 const float* __restrict__ b, float* __restrict__ out) {
    int g = blockIdx.x, o = threadIdx.x;
    float s = b[o];
    for (int k = 0; k < 9; ++k) s += af_in[g * 9 + k] * W[k * 128 + o];
    out[g * 128 + o] = s;
}

// out[g] = concat(x1, x2, af)[g] . lin3_W
__global__ __launch_bounds__(256)
void final_kernel(const float* __restrict__ x1, const float* __restrict__ x2,
                  const float* __restrict__ af, const float* __restrict__ W,
                  float* __restrict__ out) {
    int g = threadIdx.x;
    float s = 0.f;
    for (int j = 0; j < 512; ++j) s += x1[g * 512 + j] * W[j];
    for (int j = 0; j < 512; ++j) s += x2[g * 512 + j] * W[512 + j];
    for (int j = 0; j < 128; ++j) s += af[g * 128 + j] * W[1024 + j];
    out[g] = s;
}

// ---------------------------------------------------------------------------
// Host orchestration
// ---------------------------------------------------------------------------
extern "C" void kernel_launch(void* const* d_in, const int* in_sizes, int n_in,
                              void* d_out, int out_size, void* d_ws, size_t ws_size,
                              hipStream_t stream) {
    const float* x    = (const float*)d_in[0];
    const int*   ei   = (const int*)d_in[1];
    const int*   src  = ei;
    const int*   tgt  = ei + NEDGES;
    const float* addf = (const float*)d_in[3];
    // params flattened in _init_params insertion order, starting at d_in[4]:
    // per layer L (1..3), base = 4 + 13*(L-1):
    //  +0 gat_Wl +1 gat_Wr +2 gat_att +3 gat_b +4 gatn_w +5 gatn_b +6 gatn_ms
    //  +7 arma_Wi +8 arma_Wr +9 arma_b +10 arman_w +11 arman_b +12 arman_ms
    // then 43 lin1_W, 44 lin1_b, 45 lin2_W, 46 lin2_b, 47 lin3_W, 48 lin4_W, 49 lin4_b
    auto F = [&](int i) { return (const float*)d_in[i]; };

    float* ws = (float*)d_ws;
    size_t off = 0;
    auto alloc = [&](size_t n) { float* p = ws + off; off += n; return p; };
    const size_t BIG = (size_t)NNODES * 512;
    float* hgA = alloc(BIG);  float* hgB = alloc(BIG);
    float* haA = alloc(BIG);  float* haB = alloc(BIG);
    float* S0  = alloc(BIG);  float* S1  = alloc(BIG);     // XL/XI, XR/XR2
    float* H      = alloc((size_t)NNODES * 1024);
    float* logits = alloc(TOTEDGE);
    float* dinv   = alloc(NNODES);
    float* pooled = alloc((size_t)NGRAPHS * 3072);
    float* sorted = alloc((size_t)NGRAPHS * 4096);
    float* x1     = alloc((size_t)NGRAPHS * 512);
    float* x2     = alloc((size_t)NGRAPHS * 512);
    float* afb    = alloc((size_t)NGRAPHS * 128);

    auto gemm = [&](const float* A, const float* B, const float* bias, float* C,
                    int M, int N, int K) {
        int tn = (N + 15) / 16;
        int tiles = (M >> 5) * tn;                 // 32-row blocks per wave
        int blocks = (tiles + 7) / 8;
        wmma_gemm_bias_kernel<<<blocks, 256, 0, stream>>>(A, B, bias, C, M, N, K, tn);
    };

    deg_kernel<<<NGRAPHS, 256, 0, stream>>>(tgt, dinv);

    const int dims[4] = {3, 64, 128, 512};
    const float* gin = x;
    const float* ain = x;
    float* gbufs[3] = {hgA, hgB, hgA};
    float* abufs[3] = {haA, haB, haA};

    for (int L = 1; L <= 3; ++L) {
        int Ci = dims[L - 1], Co = dims[L];
        int pb = 4 + 13 * (L - 1);

        // ---- GATv2 ----
        gemm(gin, F(pb + 0), nullptr, S0, NNODES, Co, Ci);         // XL = x @ Wl
        gemm(gin, F(pb + 1), nullptr, S1, NNODES, Co, Ci);         // XR = x @ Wr
        {
            int blocks = (TOTEDGE + 7) / 8;
            edge_logits_kernel<<<blocks, 256, 0, stream>>>(S0, S1, F(pb + 2),
                                                           src, tgt, logits, Co);
        }
        softmax_stats_kernel<<<NGRAPHS, 256, 0, stream>>>(logits, tgt);
        float* gout = gbufs[L - 1];
        {
            dim3 grid(NGRAPHS, (Co + 63) / 64);
            gat_aggregate_kernel<<<grid, 256, 0, stream>>>(S0, logits, src, tgt,
                                                           F(pb + 3), gout, Co);
        }
        {
            float* dst = (L < 3) ? gout : H;
            int ldo = (L < 3) ? Co : 1024;
            dim3 grid(NGRAPHS, (Co + 255) / 256);
            graph_norm_kernel<<<grid, 256, 0, stream>>>(gout, dst, F(pb + 4),
                                                        F(pb + 5), F(pb + 6), Co, ldo, 0);
        }
        gin = gout;

        // ---- ARMA ----
        gemm(ain, F(pb + 7), nullptr,  S0, NNODES, Co, Ci);        // XI  = x @ Wi
        gemm(ain, F(pb + 8), F(pb + 9), S1, NNODES, Co, Ci);       // XR2 = x @ Wr + b
        float* aout = abufs[L - 1];
        {
            dim3 grid(NGRAPHS, (Co + 63) / 64);
            arma_aggregate_kernel<<<grid, 256, 0, stream>>>(S0, S1, dinv, src, tgt,
                                                            aout, Co);
        }
        {
            float* dst = (L < 3) ? aout : H;
            int ldo = (L < 3) ? Co : 1024;
            int coff = (L < 3) ? 0 : 512;
            dim3 grid(NGRAPHS, (Co + 255) / 256);
            graph_norm_kernel<<<grid, 256, 0, stream>>>(aout, dst, F(pb + 10),
                                                        F(pb + 11), F(pb + 12), Co, ldo, coff);
        }
        ain = aout;
    }

    pool_kernel<<<dim3(NGRAPHS, 4), 256, 0, stream>>>(H, pooled);
    sort_topk_kernel<<<NGRAPHS, 256, 0, stream>>>(H, sorted);

    gemm(pooled, F(43), F(44), x1, NGRAPHS, 512, 3072);            // lin1
    gemm(sorted, F(45), F(46), x2, NGRAPHS, 512, 4096);            // lin2
    lin4_kernel<<<NGRAPHS, 128, 0, stream>>>(addf, F(48), F(49), afb);
    final_kernel<<<1, 256, 0, stream>>>(x1, x2, afb, F(47), (float*)d_out);
}
